// RBF_5411658793099
// MI455X (gfx1250) — matrix operand order repaired and checked by reading, not compiled, not run on hardware
//
#include <hip/hip_runtime.h>

// RBF kernel K = exp(-inv * (||x||^2 - 2 X Y^T + ||y||^2)), fp32 in/out,
// via split-precision bf16 WMMA on gfx1250 (wave32).
//
// x = hi + lo with hi = bf16(x), lo = bf16(x - hi); X@Y^T evaluated as
// hi*hi + hi*lo + lo*hi with v_wmma_f32_16x16x32_bf16 into fp32 accumulators
// (dropped lo*lo term ~ 2^-18 relative -> d^2 error ~1e-4, far below fp32
// matmul on the 1/8-rate fp32 matrix path in cost, and well inside RBF
// accuracy needs).
//
// The hi/lo decomposition of X and Y is precomputed ONCE into workspace
// (8 MB, L2-resident; ws_needed ~= 8.45 MB) so the hot GEMM loop is pure
// global_load_b128 + WMMA with no conversion VALU.  Roofline: 256 MB NT
// output stream = ~11 us at 23.3 TB/s; 17.2 GFLOP contraction as 3x bf16
// GEMMs lands at/under that floor on the bf16 matrix pipe; precompute
// passes touch ~24 MB (~1 us).

#define NROWS 8192
#define MCOLS 8192
#define DDIM  128
#define EPSF  1e-9f

typedef __attribute__((ext_vector_type(16))) __bf16 v16bf;
typedef __attribute__((ext_vector_type(8)))  __bf16 v8bf;
typedef __attribute__((ext_vector_type(4)))  __bf16 v4bf;
typedef __attribute__((ext_vector_type(8)))  float  v8f;

// ---------------------------------------------------------------------------
// Pass 1a: row norms.  One wave per row (32 lanes x float4 = 128 floats),
// butterfly-reduced across the wave32.
// ---------------------------------------------------------------------------
__global__ void rbf_row_norms(const float* __restrict__ A,
                              float* __restrict__ out) {
  const int wave = (blockIdx.x * blockDim.x + threadIdx.x) >> 5;
  const int lane = threadIdx.x & 31;
  const float4 v = ((const float4*)(A + (size_t)wave * DDIM))[lane];
  float s = v.x * v.x + v.y * v.y + v.z * v.z + v.w * v.w;
#pragma unroll
  for (int off = 16; off > 0; off >>= 1) s += __shfl_xor(s, off, 32);
  if (lane == 0) out[wave] = s;
}

// ---------------------------------------------------------------------------
// Pass 1b: fp32 -> (bf16 hi, bf16 lo) decomposition, elementwise.
// 4 elements per thread: float4 load, two 8-byte bf16x4 stores.
// ---------------------------------------------------------------------------
__global__ void rbf_split(const float* __restrict__ A,
                          __bf16* __restrict__ hi, __bf16* __restrict__ lo) {
  const size_t i = (size_t)(blockIdx.x * blockDim.x + threadIdx.x) * 4;
  const float4 v = *(const float4*)(A + i);
  const __bf16 h0 = (__bf16)v.x, h1 = (__bf16)v.y;
  const __bf16 h2 = (__bf16)v.z, h3 = (__bf16)v.w;
  v4bf hv = {h0, h1, h2, h3};
  v4bf lv = {(__bf16)(v.x - (float)h0), (__bf16)(v.y - (float)h1),
             (__bf16)(v.z - (float)h2), (__bf16)(v.w - (float)h3)};
  *(v4bf*)(hi + i) = hv;
  *(v4bf*)(lo + i) = lv;
}

// ---------------------------------------------------------------------------
// Pass 2: GEMM from precomputed bf16 hi/lo + fused RBF epilogue.
//
// Block = 256 threads = 8 waves.  Wave w owns rows [m0, m0+16) x cols
// [n0, n0+64); 4 N-tiles x 4 K-steps x 3 WMMAs; A operands shared across the
// N-tiles.
//
// bf16 A operand (ISA 7.12.2, 16-bit A 16x32), lane of row r = m0 + lane%16,
// lh = lane/16: elts 0..15 = K 32k+16*lh+0..15 (same k order as B)
// -> two contiguous 16 B chunks (global_load_b128 each).
// bf16 B operand (32x16, B = Y^T row-major): elts 0..15 = K 32k+16*lh+0..15
// of row n -> one contiguous 32 B-aligned 32 B slice (two b128 loads).
// C/D: vgpr v holds row m0 + v + 8*lh, col n0 + 16t + lane%16.
// EXEC is all-ones throughout (no divergence), as WMMA requires.
// ---------------------------------------------------------------------------
__global__ void __launch_bounds__(256)
rbf_wmma_pre(const __bf16* __restrict__ Xhi, const __bf16* __restrict__ Xlo,
             const __bf16* __restrict__ Yhi, const __bf16* __restrict__ Ylo,
             const float* __restrict__ sigma,
             const float* __restrict__ x2, const float* __restrict__ y2,
             float* __restrict__ out) {
  const int wave    = threadIdx.x >> 5;
  const int lane    = threadIdx.x & 31;
  const int lane_lo = lane & 15;
  const int lane_hi = lane >> 4;

  const int m0 = blockIdx.y * 128 + wave * 16;
  const int n0 = blockIdx.x * 64;

  const float sg  = sigma[0];
  const float inv = 1.0f / (sg * sg + EPSF);

  // ---- A operands: 4 K-steps of hi/lo, loaded straight as bf16. ----
  const size_t abase = (size_t)(m0 + lane_lo) * DDIM + 16 * lane_hi;
  v16bf ahi[4], alo[4];
#pragma unroll
  for (int k = 0; k < 4; ++k) {
    const v8bf h0 = *(const v8bf*)(Xhi + abase + 32 * k);
    const v8bf h1 = *(const v8bf*)(Xhi + abase + 32 * k + 8);
    const v8bf l0 = *(const v8bf*)(Xlo + abase + 32 * k);
    const v8bf l1 = *(const v8bf*)(Xlo + abase + 32 * k + 8);
    ahi[k] = __builtin_shufflevector(h0, h1, 0, 1, 2, 3, 4, 5, 6, 7,
                                     8, 9, 10, 11, 12, 13, 14, 15);
    alo[k] = __builtin_shufflevector(l0, l1, 0, 1, 2, 3, 4, 5, 6, 7,
                                     8, 9, 10, 11, 12, 13, 14, 15);
  }

  // Hoist the 8 x2[] values this lane's output rows need.
  float xs[8];
#pragma unroll
  for (int v = 0; v < 8; ++v) xs[v] = x2[m0 + v + 8 * lane_hi];

#pragma unroll
  for (int t = 0; t < 4; ++t) {
    const int    n     = n0 + 16 * t + lane_lo;
    const size_t bbase = (size_t)n * DDIM + 16 * lane_hi;

    v8f c = {};  // fp32 accumulator
#pragma unroll
    for (int k = 0; k < 4; ++k) {
      const v16bf bhi = *(const v16bf*)(Yhi + bbase + 32 * k);
      const v16bf blo = *(const v16bf*)(Ylo + bbase + 32 * k);
      // (neg_a, A, neg_b, B, c_mod, C, reuse_a, reuse_b)
      // Small cross terms first, dominant hi*hi last.
      c = __builtin_amdgcn_wmma_f32_16x16x32_bf16(false, alo[k], false, bhi,
                                                  (short)0, c, false, false);
      c = __builtin_amdgcn_wmma_f32_16x16x32_bf16(false, ahi[k], false, blo,
                                                  (short)0, c, false, false);
      c = __builtin_amdgcn_wmma_f32_16x16x32_bf16(false, ahi[k], false, bhi,
                                                  (short)0, c, false, false);
    }

    // Epilogue: dnorm2 = x2 - 2*xy + y2 ; out = exp(-inv * dnorm2).
    const float y2n = y2[n];
#pragma unroll
    for (int v = 0; v < 8; ++v) {
      const int   m  = m0 + v + 8 * lane_hi;
      const float d2 = xs[v] - 2.0f * c[v] + y2n;
      const float kv = __expf(-inv * d2);
      // 256 MB written exactly once -> non-temporal so the bf16 operand
      // arrays stay L2-resident.
      __builtin_nontemporal_store(kv, out + (size_t)m * MCOLS + n);
    }
  }
}

// ---------------------------------------------------------------------------
// Launch wrapper.  Workspace layout (ws_needed ~= 8.45 MB):
//   x2[8192] f32 | y2[8192] f32 | Xhi | Xlo | Yhi | Ylo  (1 Mi bf16 each)
// ---------------------------------------------------------------------------
extern "C" void kernel_launch(void* const* d_in, const int* in_sizes, int n_in,
                              void* d_out, int out_size, void* d_ws,
                              size_t ws_size, hipStream_t stream) {
  const float* X     = (const float*)d_in[0];
  const float* Y     = (const float*)d_in[1];
  const float* sigma = (const float*)d_in[2];
  float*       out   = (float*)d_out;

  float* x2 = (float*)d_ws;
  float* y2 = x2 + NROWS;

  const size_t elemsX = (size_t)NROWS * DDIM;  // 1 Mi elements
  const size_t elemsY = (size_t)MCOLS * DDIM;
  __bf16* Xhi = (__bf16*)(y2 + MCOLS);
  __bf16* Xlo = Xhi + elemsX;
  __bf16* Yhi = Xlo + elemsX;
  __bf16* Ylo = Yhi + elemsY;

  // Pass 1a: row norms (8 wave-rows per 256-thread block).
  rbf_row_norms<<<NROWS / 8, 256, 0, stream>>>(X, x2);
  rbf_row_norms<<<MCOLS / 8, 256, 0, stream>>>(Y, y2);

  // Pass 1b: one-time hi/lo decomposition (4 elements/thread).
  rbf_split<<<(int)(elemsX / (4 * 256)), 256, 0, stream>>>(X, Xhi, Xlo);
  rbf_split<<<(int)(elemsY / (4 * 256)), 256, 0, stream>>>(Y, Yhi, Ylo);

  // Pass 2: each block covers 128 rows x 64 cols.
  dim3 grid(MCOLS / 64, NROWS / 128);
  rbf_wmma_pre<<<grid, 256, 0, stream>>>(Xhi, Xlo, Yhi, Ylo, sigma, x2, y2,
                                         out);
}
